// ATVP_router_wo18B_10411000726196
// MI455X (gfx1250) — matrix operand order repaired and checked by reading, not compiled, not run on hardware
//
#include <hip/hip_runtime.h>
#include <hip/hip_bf16.h>
#include <math.h>

typedef __bf16 bf16_t;
typedef __bf16 v16bf __attribute__((ext_vector_type(16)));
typedef __bf16 v8bf  __attribute__((ext_vector_type(8)));
typedef float  v8f   __attribute__((ext_vector_type(8)));

#define LDT 40  // LDS tile row stride in bf16 (80B = 5*16B: 16B-aligned, bank-conflict-free)

// ---------- helpers ----------
__device__ __forceinline__ void split2(float x, float& hi, float& lo) {
  float h = __uint_as_float(__float_as_uint(x) & 0xffff0000u);  // exact bf16 (truncated)
  hi = h; lo = x - h;                                           // residual, RNE to bf16 later
}
__device__ __forceinline__ unsigned packbf2(float a, float b) {
  union { bf16_t h[2]; unsigned u; } p;
  p.h[0] = (bf16_t)a; p.h[1] = (bf16_t)b;
  return p.u;
}
// async DMA: global -> LDS, 16B per lane, tracked by ASYNCcnt
__device__ __forceinline__ void async_b128(unsigned lds_off, const void* gaddr) {
  asm volatile("global_load_async_to_lds_b128 %0, %1, off"
               :: "v"(lds_off), "v"(gaddr) : "memory");
}
__device__ __forceinline__ void wait_async0() {
  asm volatile("s_wait_asynccnt 0x0" ::: "memory");
}
// A fragment: 16-bit A 16x32 layout. lane<16: K 0..7 (v0-3), 16..23 (v4-7); lane>=16: +8
__device__ __forceinline__ v16bf frag_a(const bf16_t* p, int kb) {
  v16bf r;
  ((v8bf*)&r)[0] = *(const v8bf*)(p + kb);
  ((v8bf*)&r)[1] = *(const v8bf*)(p + kb + 16);
  return r;
}
// B fragment: 32x16 layout, same K order as A: lane<16: K 0..7, 16..23; lane>=16: +8 (LDS holds B transposed [n][k])
__device__ __forceinline__ v16bf frag_b(const bf16_t* p, int kb) {
  v16bf r;
  ((v8bf*)&r)[0] = *(const v8bf*)(p + kb);
  ((v8bf*)&r)[1] = *(const v8bf*)(p + kb + 16);
  return r;
}

// 8 waves = 4(M) x 2(N); wave tile 32x64 = 2x4 frags; split-bf16: 3 wmma per frag pair
__device__ __forceinline__ void mma_tiles(const bf16_t* Ah, const bf16_t* Al,
                                          const bf16_t* Bh, const bf16_t* Bl,
                                          int wm, int wn, int lane, v8f acc[2][4]) {
  const int rf  = lane & 15;
  const int kbA = (lane & 16) ? 8 : 0;
  const int kbB = (lane & 16) ? 8 : 0;
  v16bf ah[2], al[2], bh[4], bl[4];
#pragma unroll
  for (int mi = 0; mi < 2; ++mi) {
    const bf16_t* p = Ah + (wm * 32 + mi * 16 + rf) * LDT;
    const bf16_t* q = Al + (wm * 32 + mi * 16 + rf) * LDT;
    ah[mi] = frag_a(p, kbA);
    al[mi] = frag_a(q, kbA);
  }
#pragma unroll
  for (int ni = 0; ni < 4; ++ni) {
    const bf16_t* p = Bh + (wn * 64 + ni * 16 + rf) * LDT;
    const bf16_t* q = Bl + (wn * 64 + ni * 16 + rf) * LDT;
    bh[ni] = frag_b(p, kbB);
    bl[ni] = frag_b(q, kbB);
  }
#pragma unroll
  for (int mi = 0; mi < 2; ++mi)
#pragma unroll
    for (int ni = 0; ni < 4; ++ni) {
      acc[mi][ni] = __builtin_amdgcn_wmma_f32_16x16x32_bf16(false, ah[mi], false, bh[ni], (short)0, acc[mi][ni], false, false);
      acc[mi][ni] = __builtin_amdgcn_wmma_f32_16x16x32_bf16(false, ah[mi], false, bl[ni], (short)0, acc[mi][ni], false, false);
      acc[mi][ni] = __builtin_amdgcn_wmma_f32_16x16x32_bf16(false, al[mi], false, bh[ni], (short)0, acc[mi][ni], false, false);
    }
}

// ---------- prep: average 7 expert weights, transpose to [N=1536][K], split bf16 hi/lo ----------
__global__ __launch_bounds__(256) void avg_transpose_kernel(
    const float* __restrict__ pW, int d, int kGlobalBase,
    bf16_t* __restrict__ WtHi, bf16_t* __restrict__ WtLo) {
  __shared__ float tile[32][33];
  const int tid = threadIdx.x;
  const int kt = blockIdx.x / 48;
  const int nt = blockIdx.x % 48;
#pragma unroll
  for (int i = 0; i < 4; ++i) {
    int e = tid + i * 256;
    int r = e >> 5, c = e & 31;
    size_t base = (size_t)(kt * 32 + r) * 1536 + nt * 32 + c;
    float s = 0.f;
#pragma unroll
    for (int ex = 0; ex < 7; ++ex) s += pW[(size_t)ex * d * 1536 + base];
    tile[r][c] = s * (1.0f / 7.0f);
  }
  __syncthreads();
#pragma unroll
  for (int i = 0; i < 2; ++i) {
    int e = tid + i * 256;
    int n = e >> 4, k2 = (e & 15) << 1;
    float h0, l0, h1, l1;
    split2(tile[k2][n], h0, l0);
    split2(tile[k2 + 1][n], h1, l1);
    size_t o = (size_t)(nt * 32 + n) * 2816 + kGlobalBase + kt * 32 + k2;
    *(unsigned*)&WtHi[o] = packbf2(h0, h1);
    *(unsigned*)&WtLo[o] = packbf2(l0, l1);
  }
}

// generic transpose + zero-pad + split (router weights)
__global__ __launch_bounds__(256) void transpose_split_kernel(
    const float* __restrict__ src, int K, int N, int dstN,
    bf16_t* __restrict__ DHi, bf16_t* __restrict__ DLo) {
  __shared__ float tile[32][33];
  const int tid = threadIdx.x;
  const int ntiles = dstN >> 5;
  const int kt = blockIdx.x / ntiles, nt = blockIdx.x % ntiles;
#pragma unroll
  for (int i = 0; i < 4; ++i) {
    int e = tid + i * 256;
    int r = e >> 5, c = e & 31;
    int col = nt * 32 + c;
    tile[r][c] = (col < N) ? src[(size_t)(kt * 32 + r) * N + col] : 0.f;
  }
  __syncthreads();
#pragma unroll
  for (int i = 0; i < 2; ++i) {
    int e = tid + i * 256;
    int n = e >> 4, k2 = (e & 15) << 1;
    float h0, l0, h1, l1;
    split2(tile[k2][n], h0, l0);
    split2(tile[k2 + 1][n], h1, l1);
    size_t o = (size_t)(nt * 32 + n) * K + kt * 32 + k2;
    *(unsigned*)&DHi[o] = packbf2(h0, h1);
    *(unsigned*)&DLo[o] = packbf2(l0, l1);
  }
}

__global__ void avg_bias_kernel(const float* __restrict__ pb0, const float* __restrict__ pb1,
                                const float* __restrict__ pbib, float* __restrict__ bbar) {
  int i = blockIdx.x * blockDim.x + threadIdx.x;
  if (i >= 3 * 1536) return;
  int s = i / 1536, j = i % 1536;
  const float* p = (s == 0) ? pb0 : ((s == 1) ? pb1 : pbib);
  float a = 0.f;
#pragma unroll
  for (int e = 0; e < 7; ++e) a += p[e * 1536 + j];
  bbar[i] = a * (1.0f / 7.0f);
}

// ---------- generic WMMA GEMM with bias (double-buffered, async B-tiles) ----------
__global__ __launch_bounds__(256) void gemm_bias_kernel(
    const float* __restrict__ A, int lda,
    const bf16_t* __restrict__ BtHi, const bf16_t* __restrict__ BtLo, int ldb,
    const float* __restrict__ bias, int biasN,
    float* __restrict__ out, int ldo, int K) {
  __shared__ bf16_t AhS[2][128 * LDT], AlS[2][128 * LDT];
  __shared__ bf16_t BhS[2][128 * LDT], BlS[2][128 * LDT];
  const int tid = threadIdx.x;
  const int lane = tid & 31, wid = tid >> 5;
  const int wm = wid & 3, wn = wid >> 2;
  const int mBase = blockIdx.y * 128;
  const int nBase = blockIdx.x * 128;
  v8f acc[2][4];
#pragma unroll
  for (int mi = 0; mi < 2; ++mi)
#pragma unroll
    for (int ni = 0; ni < 4; ++ni)
#pragma unroll
      for (int v = 0; v < 8; ++v) acc[mi][ni][v] = 0.f;

  float a0[8], a1[8];
  auto issueA = [&](int kk) {
#pragma unroll
    for (int i = 0; i < 8; ++i) {
      int e = tid + i * 256;
      int r = e >> 4, c2 = (e & 15) << 1;
      size_t o = (size_t)(mBase + r) * lda + kk + c2;
      a0[i] = A[o]; a1[i] = A[o + 1];
    }
  };
  auto storeA = [&](bf16_t* Ah, bf16_t* Al) {
#pragma unroll
    for (int i = 0; i < 8; ++i) {
      int e = tid + i * 256;
      int r = e >> 4, c2 = (e & 15) << 1;
      float h0, l0, h1, l1;
      split2(a0[i], h0, l0); split2(a1[i], h1, l1);
      *(unsigned*)&Ah[r * LDT + c2] = packbf2(h0, h1);
      *(unsigned*)&Al[r * LDT + c2] = packbf2(l0, l1);
    }
  };
  auto issueB = [&](int kk, bf16_t* Bh, bf16_t* Bl) {
#pragma unroll
    for (int j = 0; j < 2; ++j) {
      int row = (wid << 4) + (j << 3) + (lane >> 2);       // 0..127
      unsigned loff = (unsigned)(row * (LDT * 2) + ((lane & 3) << 4));
      size_t go = (size_t)(nBase + row) * ldb + kk + ((lane & 3) << 3);
      async_b128((unsigned)(uintptr_t)Bh + loff, (const void*)(BtHi + go));
      async_b128((unsigned)(uintptr_t)Bl + loff, (const void*)(BtLo + go));
    }
  };

  issueA(0);
  issueB(0, BhS[0], BlS[0]);
  storeA(AhS[0], AlS[0]);
  wait_async0();
  __syncthreads();

  const int nsteps = K >> 5;
  for (int s = 0; s < nsteps; ++s) {
    const int cur = s & 1, nxt = cur ^ 1;
    const bool has = (s + 1 < nsteps);
    if (has) { issueA((s + 1) << 5); issueB((s + 1) << 5, BhS[nxt], BlS[nxt]); }
    mma_tiles(AhS[cur], AlS[cur], BhS[cur], BlS[cur], wm, wn, lane, acc);
    if (has) storeA(AhS[nxt], AlS[nxt]);
    wait_async0();
    __syncthreads();
  }

#pragma unroll
  for (int mi = 0; mi < 2; ++mi)
#pragma unroll
    for (int ni = 0; ni < 4; ++ni) {
      int col = nBase + wn * 64 + ni * 16 + (lane & 15);
      float bv = (col < biasN) ? bias[col] : 0.f;
#pragma unroll
      for (int v = 0; v < 8; ++v) {
        int row = mBase + wm * 32 + mi * 16 + v + ((lane >> 4) << 3);
        out[(size_t)row * ldo + col] = acc[mi][ni][v] + bv;
      }
    }
}

// ---------- fused 3-segment GEMM with per-row router scaling + weighted bias ----------
__global__ __launch_bounds__(256) void combine_gemm_kernel(
    const float* __restrict__ x0, const float* __restrict__ x1, const float* __restrict__ xib,
    const bf16_t* __restrict__ WtHi, const bf16_t* __restrict__ WtLo,
    const float* __restrict__ bbar, const float* __restrict__ wgt,
    float* __restrict__ out) {
  __shared__ bf16_t AhS[2][128 * LDT], AlS[2][128 * LDT];
  __shared__ bf16_t BhS[2][128 * LDT], BlS[2][128 * LDT];
  const int tid = threadIdx.x;
  const int lane = tid & 31, wid = tid >> 5;
  const int wm = wid & 3, wn = wid >> 2;
  const int mBase = blockIdx.y * 128;
  const int nBase = blockIdx.x * 128;
  v8f acc[2][4];
#pragma unroll
  for (int mi = 0; mi < 2; ++mi)
#pragma unroll
    for (int ni = 0; ni < 4; ++ni)
#pragma unroll
      for (int v = 0; v < 8; ++v) acc[mi][ni][v] = 0.f;

  float a0[8], a1[8], sc[8];
  auto issueA = [&](int kk) {
    const float* X; int ldx, kloc, seg;
    if (kk < 1024)      { X = x0;  ldx = 1024; kloc = kk;        seg = 0; }
    else if (kk < 1792) { X = x1;  ldx = 768;  kloc = kk - 1024; seg = 1; }
    else                { X = xib; ldx = 1024; kloc = kk - 1792; seg = 2; }
#pragma unroll
    for (int i = 0; i < 8; ++i) {
      int e = tid + i * 256;
      int r = e >> 4, c2 = (e & 15) << 1;
      int row = mBase + r;
      size_t o = (size_t)row * ldx + kloc + c2;
      a0[i] = X[o]; a1[i] = X[o + 1];
      sc[i] = wgt[row * 3 + seg];
    }
  };
  auto storeA = [&](bf16_t* Ah, bf16_t* Al) {
#pragma unroll
    for (int i = 0; i < 8; ++i) {
      int e = tid + i * 256;
      int r = e >> 4, c2 = (e & 15) << 1;
      float h0, l0, h1, l1;
      split2(a0[i] * sc[i], h0, l0); split2(a1[i] * sc[i], h1, l1);
      *(unsigned*)&Ah[r * LDT + c2] = packbf2(h0, h1);
      *(unsigned*)&Al[r * LDT + c2] = packbf2(l0, l1);
    }
  };
  auto issueB = [&](int kk, bf16_t* Bh, bf16_t* Bl) {
#pragma unroll
    for (int j = 0; j < 2; ++j) {
      int row = (wid << 4) + (j << 3) + (lane >> 2);
      unsigned loff = (unsigned)(row * (LDT * 2) + ((lane & 3) << 4));
      size_t go = (size_t)(nBase + row) * 2816 + kk + ((lane & 3) << 3);
      async_b128((unsigned)(uintptr_t)Bh + loff, (const void*)(WtHi + go));
      async_b128((unsigned)(uintptr_t)Bl + loff, (const void*)(WtLo + go));
    }
  };

  issueA(0);
  issueB(0, BhS[0], BlS[0]);
  storeA(AhS[0], AlS[0]);
  wait_async0();
  __syncthreads();

  const int nsteps = 2816 >> 5;  // 88; segment boundaries (1024,1792) are multiples of 32
  for (int s = 0; s < nsteps; ++s) {
    const int cur = s & 1, nxt = cur ^ 1;
    const bool has = (s + 1 < nsteps);
    if (has) { issueA((s + 1) << 5); issueB((s + 1) << 5, BhS[nxt], BlS[nxt]); }
    mma_tiles(AhS[cur], AlS[cur], BhS[cur], BlS[cur], wm, wn, lane, acc);
    if (has) storeA(AhS[nxt], AlS[nxt]);
    wait_async0();
    __syncthreads();
  }

#pragma unroll
  for (int mi = 0; mi < 2; ++mi)
#pragma unroll
    for (int ni = 0; ni < 4; ++ni) {
      int col = nBase + wn * 64 + ni * 16 + (lane & 15);
      float b0c = bbar[col], b1c = bbar[1536 + col], b2c = bbar[3072 + col];
#pragma unroll
      for (int v = 0; v < 8; ++v) {
        int row = mBase + wm * 32 + mi * 16 + v + ((lane >> 4) << 3);
        float w0 = wgt[row * 3 + 0], w1 = wgt[row * 3 + 1], w2 = wgt[row * 3 + 2];
        out[(size_t)row * 1536 + col] = acc[mi][ni][v] + w0 * b0c + w1 * b1c + w2 * b2c;
      }
    }
}

// ---------- batchnorm (two-stage, deterministic) ----------
__global__ void bn_partial_kernel(const float* __restrict__ X, int ld, int C, int rows,
                                  float* __restrict__ part) {
  int col = blockIdx.x * 128 + threadIdx.x;
  if (col >= C) return;
  int chunk = rows >> 5;
  int r0 = blockIdx.y * chunk;
  float s = 0.f, q = 0.f;
  for (int r = r0; r < r0 + chunk; ++r) {
    float v = X[(size_t)r * ld + col];
    s += v; q += v * v;
  }
  part[blockIdx.y * C + col] = s;
  part[32 * C + blockIdx.y * C + col] = q;
}
__global__ void bn_finalize_kernel(const float* __restrict__ part, int C, float invRows,
                                   float* __restrict__ meanOut, float* __restrict__ istdOut) {
  int col = blockIdx.x * 128 + threadIdx.x;
  if (col >= C) return;
  float s = 0.f, q = 0.f;
  for (int c2 = 0; c2 < 32; ++c2) { s += part[c2 * C + col]; q += part[32 * C + c2 * C + col]; }
  float m = s * invRows;
  float v = q * invRows - m * m;  // biased var (matches jnp.var)
  meanOut[col] = m;
  istdOut[col] = rsqrtf(v + 1e-5f);
}
__global__ void bn_relu_kernel(float* __restrict__ X, int C, long long total,
                               const float* __restrict__ mean, const float* __restrict__ istd,
                               const float* __restrict__ g, const float* __restrict__ b) {
  long long idx = (long long)blockIdx.x * blockDim.x + threadIdx.x;
  if (idx >= total) return;
  int c = (int)(idx % C);
  float y = g[c] * (X[idx] - mean[c]) * istd[c] + b[c];
  X[idx] = fmaxf(y, 0.f);
}

// ---------- router tail: BN2 + tanh + linear(100,3) + sigmoid + softmax ----------
__global__ void router_tail_kernel(const float* __restrict__ G2,
                                   const float* __restrict__ mean2, const float* __restrict__ istd2,
                                   const float* __restrict__ g2, const float* __restrict__ bt2,
                                   const float* __restrict__ rw3, const float* __restrict__ rb3,
                                   float* __restrict__ wgt, int rows) {
  int b = blockIdx.x * blockDim.x + threadIdx.x;
  if (b >= rows) return;
  float a0 = rb3[0], a1 = rb3[1], a2 = rb3[2];
  for (int c = 0; c < 100; ++c) {
    float v = G2[(size_t)b * 128 + c];
    v = g2[c] * (v - mean2[c]) * istd2[c] + bt2[c];
    v = tanhf(v);
    a0 += v * rw3[c * 3 + 0];
    a1 += v * rw3[c * 3 + 1];
    a2 += v * rw3[c * 3 + 2];
  }
  float s0 = 1.f / (1.f + expf(-a0));
  float s1 = 1.f / (1.f + expf(-a1));
  float s2 = 1.f / (1.f + expf(-a2));
  float mx = fmaxf(s0, fmaxf(s1, s2));
  float e0 = expf(s0 - mx), e1 = expf(s1 - mx), e2 = expf(s2 - mx);
  float inv = 1.f / (e0 + e1 + e2);
  wgt[b * 3 + 0] = e0 * inv;
  wgt[b * 3 + 1] = e1 * inv;
  wgt[b * 3 + 2] = e2 * inv;
}

// ---------- final L2 row-normalize (in place on d_out) ----------
__global__ __launch_bounds__(256) void rownorm_kernel(float* __restrict__ out) {
  __shared__ float red[256];
  int r = blockIdx.x, t = threadIdx.x;
  float vals[6];
  float s = 0.f;
#pragma unroll
  for (int i = 0; i < 6; ++i) {
    vals[i] = out[(size_t)r * 1536 + t + i * 256];
    s += vals[i] * vals[i];
  }
  red[t] = s;
  __syncthreads();
  for (int k = 128; k > 0; k >>= 1) {
    if (t < k) red[t] += red[t + k];
    __syncthreads();
  }
  float scale = 1.f / fmaxf(sqrtf(red[0]), 1e-12f);
#pragma unroll
  for (int i = 0; i < 6; ++i) out[(size_t)r * 1536 + t + i * 256] = vals[i] * scale;
}

// ---------- launch ----------
extern "C" void kernel_launch(void* const* d_in, const int* in_sizes, int n_in,
                              void* d_out, int out_size, void* d_ws, size_t ws_size,
                              hipStream_t stream) {
  (void)in_sizes; (void)n_in; (void)out_size; (void)ws_size;
  const float* x0   = (const float*)d_in[0];
  const float* x1   = (const float*)d_in[1];
  const float* xib  = (const float*)d_in[2];
  const float* pW0  = (const float*)d_in[3];
  const float* pb0  = (const float*)d_in[4];
  const float* pW1  = (const float*)d_in[5];
  const float* pb1  = (const float*)d_in[6];
  const float* pWib = (const float*)d_in[7];
  const float* pbib = (const float*)d_in[8];
  const float* rw1  = (const float*)d_in[9];
  const float* rb1  = (const float*)d_in[10];
  const float* rg1  = (const float*)d_in[11];
  const float* rbt1 = (const float*)d_in[12];
  const float* rw2  = (const float*)d_in[13];
  const float* rb2  = (const float*)d_in[14];
  const float* rg2  = (const float*)d_in[15];
  const float* rbt2 = (const float*)d_in[16];
  const float* rw3  = (const float*)d_in[17];
  const float* rb3  = (const float*)d_in[18];
  float* out = (float*)d_out;

  char* ws = (char*)d_ws;
  size_t off = 0;
  auto alloc = [&](size_t bytes) -> char* {
    char* p = ws + off;
    off += (bytes + 255) & ~(size_t)255;
    return p;
  };
  bf16_t* WtHi  = (bf16_t*)alloc(1536ull * 2816 * 2);
  bf16_t* WtLo  = (bf16_t*)alloc(1536ull * 2816 * 2);
  float*  bbar  = (float*)alloc(3ull * 1536 * 4);
  bf16_t* Rt1Hi = (bf16_t*)alloc(512ull * 1024 * 2);
  bf16_t* Rt1Lo = (bf16_t*)alloc(512ull * 1024 * 2);
  bf16_t* Rt2Hi = (bf16_t*)alloc(128ull * 512 * 2);
  bf16_t* Rt2Lo = (bf16_t*)alloc(128ull * 512 * 2);
  float*  G1    = (float*)alloc(8192ull * 512 * 4);
  float*  G2    = (float*)alloc(8192ull * 128 * 4);
  float*  part1 = (float*)alloc(64ull * 512 * 4);
  float*  mean1 = (float*)alloc(512ull * 4);
  float*  istd1 = (float*)alloc(512ull * 4);
  float*  part2 = (float*)alloc(64ull * 128 * 4);
  float*  mean2 = (float*)alloc(128ull * 4);
  float*  istd2 = (float*)alloc(128ull * 4);
  float*  wgt   = (float*)alloc(8192ull * 3 * 4);

  // 1) prep averaged / transposed split-bf16 weights
  avg_transpose_kernel<<<dim3(32 * 48), 256, 0, stream>>>(pW0, 1024, 0, WtHi, WtLo);
  avg_transpose_kernel<<<dim3(24 * 48), 256, 0, stream>>>(pW1, 768, 1024, WtHi, WtLo);
  avg_transpose_kernel<<<dim3(32 * 48), 256, 0, stream>>>(pWib, 1024, 1792, WtHi, WtLo);
  avg_bias_kernel<<<18, 256, 0, stream>>>(pb0, pb1, pbib, bbar);
  transpose_split_kernel<<<32 * 16, 256, 0, stream>>>(rw1, 1024, 512, 512, Rt1Hi, Rt1Lo);
  transpose_split_kernel<<<16 * 4, 256, 0, stream>>>(rw2, 512, 100, 128, Rt2Hi, Rt2Lo);

  // 2) router layer 1: GEMM + BN + relu
  gemm_bias_kernel<<<dim3(4, 64), 256, 0, stream>>>(x0, 1024, Rt1Hi, Rt1Lo, 1024, rb1, 512, G1, 512, 1024);
  bn_partial_kernel<<<dim3(4, 32), 128, 0, stream>>>(G1, 512, 512, 8192, part1);
  bn_finalize_kernel<<<4, 128, 0, stream>>>(part1, 512, 1.f / 8192.f, mean1, istd1);
  bn_relu_kernel<<<(8192 * 512 + 255) / 256, 256, 0, stream>>>(G1, 512, 8192ll * 512, mean1, istd1, rg1, rbt1);

  // 3) router layer 2 (N padded 100->128) + BN + tail
  gemm_bias_kernel<<<dim3(1, 64), 256, 0, stream>>>(G1, 512, Rt2Hi, Rt2Lo, 512, rb2, 100, G2, 128, 512);
  bn_partial_kernel<<<dim3(1, 32), 128, 0, stream>>>(G2, 128, 100, 8192, part2);
  bn_finalize_kernel<<<1, 128, 0, stream>>>(part2, 100, 1.f / 8192.f, mean2, istd2);
  router_tail_kernel<<<32, 256, 0, stream>>>(G2, mean2, istd2, rg2, rbt2, rw3, rb3, wgt, 8192);

  // 4) fused expert GEMM + weighted combine, then 5) L2 normalize
  combine_gemm_kernel<<<dim3(12, 64), 256, 0, stream>>>(x0, x1, xib, WtHi, WtLo, bbar, wgt, out);
  rownorm_kernel<<<8192, 256, 0, stream>>>(out);
}
